// WorldLatticeProjector_34342558499433
// MI455X (gfx1250) — matrix operand order repaired and checked by reading, not compiled, not run on hardware
//
#include <hip/hip_runtime.h>

typedef _Float16 v16h __attribute__((ext_vector_type(16)));
typedef _Float16 v8h  __attribute__((ext_vector_type(8)));
typedef _Float16 v4h  __attribute__((ext_vector_type(4)));
typedef float    v8f  __attribute__((ext_vector_type(8)));

#define NB   8      // batches
#define NT   16
#define ND   64
#define NPIX 4096   // HP*WP
#define KL   32     // lattice edge
#define KK   1024   // KL*KL  (GEMM N)
#define TD   1024   // T*D    (GEMM M)
#define BK   64     // k-chunk (pixels per chunk), 2 WMMA k-steps of 32
#define TM   128    // C tile rows per workgroup
#define TN   128    // C tile cols per workgroup
#define SA   72     // A LDS row stride in halves (rows 144B, 16B-aligned)
#define SB   72     // S LDS column stride in halves (column-major: K contiguous)
#define EPSF 1e-6f

__device__ __forceinline__ void prefetch_chunk(const float* __restrict__ Ag,
                                               int c0, int tid, float4* pref) {
  #pragma unroll
  for (int i = 0; i < 8; ++i) {    // 128x64 floats = 2048 float4 -> 8 per thread
    int f   = tid + i * 256;
    int row = f >> 4;              // 16 float4 per row
    int cf  = f & 15;
    pref[i] = ((const float4*)(Ag + (size_t)row * NPIX + c0))[cf];
  }
}

__device__ __forceinline__ v16h frag16(const _Float16* p, int step) {
  v8h lo = *(const v8h*)(p);
  v8h hi = *(const v8h*)(p + step);
  return __builtin_shufflevector(lo, hi,
         0,1,2,3,4,5,6,7,8,9,10,11,12,13,14,15);
}

// world[b] (TD x KK) = feats[b] (TD x NPIX) * S[b] (NPIX x KK), then /= max(colsum, eps)
__global__ __launch_bounds__(256) void splat_gemm_kernel(
    const float* __restrict__ feats,
    const float* __restrict__ coord,
    float* __restrict__ world) {
  __shared__ _Float16 Alds[TM * SA];   // 128 rows x 64 K (f16), row-major
  __shared__ _Float16 Slds[TN * SB];   // 128 cols x 64 K (f16), column-major (K contiguous)
  __shared__ float    colsum[TN];      // running fp32 column sums (== wsum slice)

  const int tid = threadIdx.x;
  const int bid = blockIdx.x;
  const int b   = bid >> 6;          // 64 tiles per batch
  const int tm  = (bid >> 3) & 7;    // row tile
  const int tn  = bid & 7;           // col tile

  const float* Ag = feats + (size_t)b * TD * NPIX + (size_t)tm * TM * NPIX;
  const float* Cg = coord + (size_t)b * NPIX * 2;

  const int lane = tid & 31;
  const int wave = tid >> 5;
  const int wr   = wave >> 1;        // 0..3: row group (2 strips of 16 -> 32 rows)
  const int wc   = wave & 1;         // 0..1: col group (4 tiles of 16 -> 64 cols)
  const int m16  = lane & 15;
  const int khi  = lane >> 4;        // lane-half of the fragment K layout

  v8f acc[2][4] = {};                // [row strip][col tile] 16x16 f32 sub-tiles

  if (tid < TN) colsum[tid] = 0.f;

  const int colbase = tn * TN;

  // software pipeline: chunk 0 prefetched before the loop
  float4 pref[8];
  prefetch_chunk(Ag, 0, tid, pref);

  for (int c0 = 0; c0 < NPIX; c0 += BK) {
    __syncthreads();  // WAR: previous chunk fully consumed before overwrite

    // ---- phase 1: zero S tile, commit prefetched A (f32->f16), prefetch next ----
    {
      float4* S4 = (float4*)Slds;    // 128*72 halves = 1152 float4
      #pragma unroll
      for (int i = 0; i < 5; ++i) {
        int idx = tid + i * 256;
        if (idx < (TN * SB) / 8) S4[idx] = make_float4(0.f, 0.f, 0.f, 0.f);
      }
    }
    #pragma unroll
    for (int i = 0; i < 8; ++i) {
      int f   = tid + i * 256;
      int row = f >> 4;
      int cf  = f & 15;
      float4 v = pref[i];
      v4h h = { (_Float16)v.x, (_Float16)v.y, (_Float16)v.z, (_Float16)v.w };
      *((v4h*)(Alds + row * SA + cf * 4)) = h;
    }
    if (c0 + BK < NPIX)              // issue next chunk's global loads now;
      prefetch_chunk(Ag, c0 + BK, tid, pref);  // waited only next iteration
    __syncthreads();

    // ---- phase 2: build splat matrix (one pixel per thread, column-major S) ----
    if (tid < BK) {
      int p = c0 + tid;
      float x = Cg[p * 2 + 0];
      float y = Cg[p * 2 + 1];
      float gx = (x + 15.0f) * (1.0f / 30.0f) * 31.0f;
      float gy = (y + 15.0f) * (1.0f / 30.0f) * 31.0f;
      float fx0 = floorf(gx), fy0 = floorf(gy);
      int   x0 = (int)fx0,    y0 = (int)fy0;
      float wx1 = gx - fx0,   wy1 = gy - fy0;
      float wx0 = 1.f - wx1,  wy0 = 1.f - wy1;
      #define SPLAT(cx, cy, w) do {                                   \
        if ((unsigned)(cx) < KL && (unsigned)(cy) < KL) {             \
          int lc = (cy) * KL + (cx) - colbase;                        \
          if ((unsigned)lc < TN) Slds[lc * SB + tid] = (_Float16)(w); \
        } } while (0)
      SPLAT(x0,     y0,     wx0 * wy0);
      SPLAT(x0 + 1, y0,     wx1 * wy0);
      SPLAT(x0,     y0 + 1, wx0 * wy1);
      SPLAT(x0 + 1, y0 + 1, wx1 * wy1);
      #undef SPLAT
    }
    __syncthreads();

    // ---- phase 3a: accumulate fp32 column sums (thread per column) ----
    if (tid < TN) {
      float s = 0.f;
      #pragma unroll 8
      for (int k = 0; k < BK; ++k) s += (float)Slds[tid * SB + k];
      colsum[tid] += s;
    }

    // ---- phase 3b: WMMA f16 (EXEC all-ones) ----
    // Wave grid 4x2: rows [wr*32, +32) as two 16-strips, cols [wc*64, +64) as
    // four 16-tiles. Per 32-K step: 2 A-frags + 4 B-frags feed 8 WMMAs.
    // A 16x32 f16 frag: lane(m16,khi); elems 0..7 -> K=ks+8*khi+e,
    //                   elems 8..15 -> K=ks+16+8*khi+(e-8)
    // B 32x16 f16 frag: lane(m16,khi); same K layout as A, col=m16
    const _Float16* Abase = Alds + (wr * 32 + m16) * SA + 8 * khi;
    const _Float16* Bbase = Slds + (wc * 64 + m16) * SB + 8 * khi;
    #pragma unroll
    for (int ks = 0; ks < BK; ks += 32) {
      v16h a0 = frag16(Abase + ks, 16);
      v16h a1 = frag16(Abase + 16 * SA + ks, 16);
      v16h bf[4];
      #pragma unroll
      for (int jc = 0; jc < 4; ++jc)
        bf[jc] = frag16(Bbase + jc * (16 * SB) + ks, 16);
      #pragma unroll
      for (int jc = 0; jc < 4; ++jc) {
        acc[0][jc] = __builtin_amdgcn_wmma_f32_16x16x32_f16(
            false, a0, false, bf[jc], (short)0, acc[0][jc], false, false);
        acc[1][jc] = __builtin_amdgcn_wmma_f32_16x16x32_f16(
            false, a1, false, bf[jc], (short)0, acc[1][jc], false, false);
      }
    }
  }
  __syncthreads();

  // ---- epilogue: divide by max(wsum, eps), store fp32 ----
  float* Wg = world + (size_t)b * TD * KK + (size_t)(tm * TM) * KK + colbase;
  #pragma unroll
  for (int ir = 0; ir < 2; ++ir) {
    #pragma unroll
    for (int jc = 0; jc < 4; ++jc) {
      int lc = wc * 64 + jc * 16 + m16;
      float inv = 1.0f / fmaxf(colsum[lc], EPSF);
      #pragma unroll
      for (int v = 0; v < 8; ++v) {
        int row = wr * 32 + ir * 16 + v + 8 * khi;  // C VGPR v: M=v / v+8 per half
        Wg[(size_t)row * KK + lc] = acc[ir][jc][v] * inv;
      }
    }
  }
}

// weights output: fp32-exact wsum[b, cell] broadcast over T
__global__ __launch_bounds__(256) void weights_kernel(
    const float* __restrict__ coord, float* __restrict__ outw) {
  __shared__ float wsum[KK];
  const int b = blockIdx.x;
  const int tid = threadIdx.x;
  for (int i = tid; i < KK; i += 256) wsum[i] = 0.f;
  __syncthreads();
  const float* Cg = coord + (size_t)b * NPIX * 2;
  for (int p = tid; p < NPIX; p += 256) {
    float x = Cg[p * 2 + 0];
    float y = Cg[p * 2 + 1];
    float gx = (x + 15.0f) * (1.0f / 30.0f) * 31.0f;
    float gy = (y + 15.0f) * (1.0f / 30.0f) * 31.0f;
    float fx0 = floorf(gx), fy0 = floorf(gy);
    int   x0 = (int)fx0,    y0 = (int)fy0;
    float wx1 = gx - fx0,   wy1 = gy - fy0;
    float wx0 = 1.f - wx1,  wy0 = 1.f - wy1;
    #define ACCW(cx, cy, w) do {                                      \
      if ((unsigned)(cx) < KL && (unsigned)(cy) < KL)                 \
        atomicAdd(&wsum[(cy) * KL + (cx)], (w));                      \
      } while (0)
    ACCW(x0,     y0,     wx0 * wy0);
    ACCW(x0 + 1, y0,     wx1 * wy0);
    ACCW(x0,     y0 + 1, wx0 * wy1);
    ACCW(x0 + 1, y0 + 1, wx1 * wy1);
    #undef ACCW
  }
  __syncthreads();
  float* Og = outw + (size_t)b * NT * KK;
  for (int i = tid; i < KK; i += 256) {
    float v = wsum[i];
    #pragma unroll
    for (int t = 0; t < NT; ++t) Og[(size_t)t * KK + i] = v;
  }
}

extern "C" void kernel_launch(void* const* d_in, const int* in_sizes, int n_in,
                              void* d_out, int out_size, void* d_ws, size_t ws_size,
                              hipStream_t stream) {
  (void)in_sizes; (void)n_in; (void)out_size; (void)d_ws; (void)ws_size;
  const float* feats = (const float*)d_in[0];   // (B,T,D,HP,WP) f32
  const float* coord = (const float*)d_in[1];   // (B,HP,WP,2)  f32
  float* world = (float*)d_out;                              // (B,T,D,K,K)
  float* outw  = world + (size_t)NB * NT * ND * KK;          // (B,T,1,K,K)
  splat_gemm_kernel<<<NB * 8 * 8, 256, 0, stream>>>(feats, coord, world);
  weights_kernel<<<NB, 256, 0, stream>>>(coord, outw);
}
